// ModalityHarmonizer_7421703487988
// MI455X (gfx1250) — matrix operand order repaired and checked by reading, not compiled, not run on hardware
//
// ModalityHarmonizer for MI455X (gfx1250, CDNA5, wave32).
// GEMM-shaped math via v_wmma_f32_16x16x32_f16 with register blocking:
// each wave computes a 16x64 C tile (1 A-frag x 4 B-frags -> 4 WMMAs / K-step),
// block of 4 waves covers 64x64. Guard-free hot loops (shapes padded to
// 64x64x32 tiling), float4 (global_load_b128) fragment loads. Attention scores
// materialized in workspace (fits 192MB L2); selective scan is VALU + shfl.
#include <hip/hip_runtime.h>
#include <cstdint>

typedef __attribute__((ext_vector_type(16))) _Float16 v16h;
typedef __attribute__((ext_vector_type(8)))  float    v8f;

namespace {
constexpr int Bc = 2, Lc = 2048, Hc = 4, DHc = 64, Ec = 256;
constexpr int DHALFc = 128;
constexpr int MT = Bc * Lc;          // 4096 token rows
constexpr long long LL = (long long)Lc * Lc;
constexpr long long LD = (long long)Lc * DHc;
}

// A-fragment, 16-bit A 16x32: lane(l16)=row, K = {16h+0..15} contiguous
__device__ __forceinline__ v16h load_frag_a(const float* __restrict__ p)
{
  float4 x0 = *(const float4*)(p);
  float4 x1 = *(const float4*)(p + 4);
  float4 x2 = *(const float4*)(p + 8);
  float4 x3 = *(const float4*)(p + 12);
  v16h f;
  f[0]=(_Float16)x0.x; f[1]=(_Float16)x0.y; f[2]=(_Float16)x0.z; f[3]=(_Float16)x0.w;
  f[4]=(_Float16)x1.x; f[5]=(_Float16)x1.y; f[6]=(_Float16)x1.z; f[7]=(_Float16)x1.w;
  f[8]=(_Float16)x2.x; f[9]=(_Float16)x2.y; f[10]=(_Float16)x2.z; f[11]=(_Float16)x2.w;
  f[12]=(_Float16)x3.x; f[13]=(_Float16)x3.y; f[14]=(_Float16)x3.z; f[15]=(_Float16)x3.w;
  return f;
}

// B-fragment (NT): lane(l16)=col, K = {16h+0..15} contiguous in W's row
__device__ __forceinline__ v16h load_frag16(const float* __restrict__ p)
{
  float4 x0 = *(const float4*)(p);
  float4 x1 = *(const float4*)(p + 4);
  float4 x2 = *(const float4*)(p + 8);
  float4 x3 = *(const float4*)(p + 12);
  v16h f;
  f[0]=(_Float16)x0.x; f[1]=(_Float16)x0.y; f[2]=(_Float16)x0.z; f[3]=(_Float16)x0.w;
  f[4]=(_Float16)x1.x; f[5]=(_Float16)x1.y; f[6]=(_Float16)x1.z; f[7]=(_Float16)x1.w;
  f[8]=(_Float16)x2.x; f[9]=(_Float16)x2.y; f[10]=(_Float16)x2.z; f[11]=(_Float16)x2.w;
  f[12]=(_Float16)x3.x; f[13]=(_Float16)x3.y; f[14]=(_Float16)x3.z; f[15]=(_Float16)x3.w;
  return f;
}

// B-fragment (NN): column access, stride ldb; coalesced across lanes
__device__ __forceinline__ v16h load_frag_bnn(const float* __restrict__ p, int ldb)
{
  v16h f;
#pragma unroll
  for (int t = 0; t < 16; ++t) f[t] = (_Float16)p[(long long)t * ldb];
  return f;
}

#define WMMA_F16(af, bf, acc) \
  __builtin_amdgcn_wmma_f32_16x16x32_f16(false, (af), false, (bf), (short)0, (acc), false, false)

// ---------------------------------------------------------------------------
// WMMA GEMM, C[M,N] = A[M,K] * W[N,K]^T (+bias). Batched via grid.z strides.
// REQUIRES: M % 64 == 0, N % 64 == 0, K % 32 == 0 (host guarantees/pads).
// grid = (N/64, M/64, batch); block = 128 (4 waves); wave tile = 16x64.
// ---------------------------------------------------------------------------
__global__ void wmma_gemm_nt(const float* __restrict__ A, const float* __restrict__ W,
                             const float* __restrict__ bias, float* __restrict__ C,
                             int K, int lda, int ldw, int ldc,
                             long long sA, long long sW, long long sC)
{
  const int bz = blockIdx.z;
  A += (long long)bz * sA;  W += (long long)bz * sW;  C += (long long)bz * sC;
  const int lane = threadIdx.x & 31;
  const int wave = threadIdx.x >> 5;
  const int m0 = blockIdx.y * 64 + wave * 16;
  const int n0 = blockIdx.x * 64;
  const int hl  = lane >> 4;
  const int l16 = lane & 15;
  const float* Ar = A + (long long)(m0 + l16) * lda + 16 * hl;
  const float* Wr = W + (long long)(n0 + l16) * ldw + 16 * hl;
  const long long wstep = (long long)16 * ldw;
  v8f acc0 = {}, acc1 = {}, acc2 = {}, acc3 = {};
  for (int kk = 0; kk < K; kk += 32) {
    __builtin_prefetch(Ar + kk + 32, 0, 1);
    __builtin_prefetch(Wr + kk + 32, 0, 1);
    v16h af = load_frag_a(Ar + kk);
    v16h b0 = load_frag16(Wr + kk);
    v16h b1 = load_frag16(Wr + wstep + kk);
    v16h b2 = load_frag16(Wr + 2 * wstep + kk);
    v16h b3 = load_frag16(Wr + 3 * wstep + kk);
    acc0 = WMMA_F16(af, b0, acc0);
    acc1 = WMMA_F16(af, b1, acc1);
    acc2 = WMMA_F16(af, b2, acc2);
    acc3 = WMMA_F16(af, b3, acc3);
  }
  const float bv0 = bias ? bias[n0 + l16]      : 0.f;
  const float bv1 = bias ? bias[n0 + 16 + l16] : 0.f;
  const float bv2 = bias ? bias[n0 + 32 + l16] : 0.f;
  const float bv3 = bias ? bias[n0 + 48 + l16] : 0.f;
#pragma unroll
  for (int r = 0; r < 8; ++r) {                // C/D: row = r + 8*hl, col = l16
    float* Cr = C + (long long)(m0 + r + 8 * hl) * ldc + n0 + l16;
    Cr[0]  = acc0[r] + bv0;
    Cr[16] = acc1[r] + bv1;
    Cr[32] = acc2[r] + bv2;
    Cr[48] = acc3[r] + bv3;
  }
}

// C[M,N] = A[M,K] * Bm[K,N]  (Bm row-major over K). Same tiling requirements.
__global__ void wmma_gemm_nn(const float* __restrict__ A, const float* __restrict__ Bm,
                             float* __restrict__ C,
                             int K, int lda, int ldb, int ldc,
                             long long sA, long long sB, long long sC)
{
  const int bz = blockIdx.z;
  A += (long long)bz * sA;  Bm += (long long)bz * sB;  C += (long long)bz * sC;
  const int lane = threadIdx.x & 31;
  const int wave = threadIdx.x >> 5;
  const int m0 = blockIdx.y * 64 + wave * 16;
  const int n0 = blockIdx.x * 64;
  const int hl  = lane >> 4;
  const int l16 = lane & 15;
  const float* Ar = A + (long long)(m0 + l16) * lda + 16 * hl;
  const float* Br = Bm + (long long)(16 * hl) * ldb + (n0 + l16);
  v8f acc0 = {}, acc1 = {}, acc2 = {}, acc3 = {};
  for (int kk = 0; kk < K; kk += 32) {
    __builtin_prefetch(Ar + kk + 32, 0, 1);
    v16h af = load_frag_a(Ar + kk);
    const float* Bk = Br + (long long)kk * ldb;
    v16h b0 = load_frag_bnn(Bk, ldb);
    v16h b1 = load_frag_bnn(Bk + 16, ldb);
    v16h b2 = load_frag_bnn(Bk + 32, ldb);
    v16h b3 = load_frag_bnn(Bk + 48, ldb);
    acc0 = WMMA_F16(af, b0, acc0);
    acc1 = WMMA_F16(af, b1, acc1);
    acc2 = WMMA_F16(af, b2, acc2);
    acc3 = WMMA_F16(af, b3, acc3);
  }
#pragma unroll
  for (int r = 0; r < 8; ++r) {
    float* Cr = C + (long long)(m0 + r + 8 * hl) * ldc + n0 + l16;
    Cr[0]  = acc0[r];
    Cr[16] = acc1[r];
    Cr[32] = acc2[r];
    Cr[48] = acc3[r];
  }
}

// ---------------------------------------------------------------------------
// Row softmax (in place): one block per row; grid.y = batch.
// ---------------------------------------------------------------------------
__global__ void softmax_rows(float* __restrict__ S, int cols, float scale,
                             long long srow, long long sbatch)
{
  float* p = S + (long long)blockIdx.y * sbatch + (long long)blockIdx.x * srow;
  __shared__ float red[256];
  const int tid = threadIdx.x;
  float m = -3.4e38f;
  for (int c = tid; c < cols; c += 256) m = fmaxf(m, p[c] * scale);
  red[tid] = m; __syncthreads();
  for (int s = 128; s > 0; s >>= 1) { if (tid < s) red[tid] = fmaxf(red[tid], red[tid+s]); __syncthreads(); }
  m = red[0]; __syncthreads();
  float sum = 0.f;
  for (int c = tid; c < cols; c += 256) { float e = expf(p[c]*scale - m); p[c] = e; sum += e; }
  red[tid] = sum; __syncthreads();
  for (int s = 128; s > 0; s >>= 1) { if (tid < s) red[tid] += red[tid+s]; __syncthreads(); }
  float inv = 1.f / red[0];
  for (int c = tid; c < cols; c += 256) p[c] *= inv;
}

// Column softmax, result written TRANSPOSED: PT[j,i] = softmax_i(S[i,j]).
__global__ void softmax_cols_T(const float* __restrict__ S, float* __restrict__ PT,
                               int rows, int cols, long long sbin, long long sbout)
{
  const float* sp = S + (long long)blockIdx.y * sbin;
  float* op = PT + (long long)blockIdx.y * sbout + (long long)blockIdx.x * rows;
  const int j = blockIdx.x, tid = threadIdx.x;
  __shared__ float red[256];
  float m = -3.4e38f;
  for (int i = tid; i < rows; i += 256) m = fmaxf(m, sp[(long long)i * cols + j]);
  red[tid] = m; __syncthreads();
  for (int s = 128; s > 0; s >>= 1) { if (tid < s) red[tid] = fmaxf(red[tid], red[tid+s]); __syncthreads(); }
  m = red[0]; __syncthreads();
  float sum = 0.f;
  for (int i = tid; i < rows; i += 256) { float e = expf(sp[(long long)i*cols + j] - m); op[i] = e; sum += e; }
  red[tid] = sum; __syncthreads();
  for (int s = 128; s > 0; s >>= 1) { if (tid < s) red[tid] += red[tid+s]; __syncthreads(); }
  float inv = 1.f / red[0];
  for (int i = tid; i < rows; i += 256) op[i] *= inv;
}

// ---------------------------------------------------------------------------
// QKV split + RoPE:  qkv (MT,3E) interleaved (h,d,comp) -> Q,K,V (B,H,L,DH)
// ---------------------------------------------------------------------------
__global__ void rope_split(const float* __restrict__ qkv, const float* __restrict__ pe,
                           float* __restrict__ Q, float* __restrict__ K, float* __restrict__ V)
{
  int idx = blockIdx.x * 256 + threadIdx.x;        // ((b*H+h)*L+l)*DH+d
  int d = idx & (DHc - 1);
  int rest = idx >> 6;
  int l = rest & (Lc - 1);
  int bh = rest >> 11;
  int h = bh & (Hc - 1);
  int b = bh >> 2;
  long long row = (long long)b * Lc + l;
  long long base = row * (3 * Ec) + (long long)(h * DHc + d) * 3;
  float q = qkv[base + 0], k = qkv[base + 1], v = qkv[base + 2];
  long long pbase = ((long long)(d & 1 ? base - 3 : base + 3));
  float qp = qkv[pbase + 0], kp = qkv[pbase + 1];
  float rhq = (d & 1) ? qp : -qp;                  // rotate_half
  float rhk = (d & 1) ? kp : -kp;
  float p0 = pe[((long long)b * Lc + l) * DHc + d];
  float p1 = pe[((long long)(Bc + b) * Lc + l) * DHc + d];
  Q[idx] = q * p0 + rhq * p1;
  K[idx] = k * p0 + rhk * p1;
  V[idx] = v;
}

__global__ void split_heads_scale(const float* __restrict__ in, float* __restrict__ out, float scale)
{
  int idx = blockIdx.x * 256 + threadIdx.x;
  int d = idx & (DHc - 1);
  int rest = idx >> 6;
  int l = rest & (Lc - 1);
  int bh = rest >> 11;
  int h = bh & (Hc - 1);
  int b = bh >> 2;
  out[idx] = in[((long long)b * Lc + l) * Ec + h * DHc + d] * scale;
}

__global__ void merge_heads(const float* __restrict__ in, float* __restrict__ out)
{
  int idx = blockIdx.x * 256 + threadIdx.x;
  int d = idx & (DHc - 1);
  int rest = idx >> 6;
  int l = rest & (Lc - 1);
  int bh = rest >> 11;
  int h = bh & (Hc - 1);
  int b = bh >> 2;
  out[((long long)b * Lc + l) * Ec + h * DHc + d] = in[idx];
}

// out[r, colOff + c] = src[r*srcLd + c]  (c < srcC)
__global__ void copy_cols(float* __restrict__ out, const float* __restrict__ src,
                          int rows, int srcC, int srcLd, int outC, int colOff)
{
  int idx = blockIdx.x * 256 + threadIdx.x;
  if (idx >= rows * srcC) return;
  int r = idx / srcC, c = idx - r * srcC;
  out[(long long)r * outC + colOff + c] = src[(long long)r * srcLd + c];
}

// out (outR,outC) = src (srcR,srcC @ srcLd), zero-padded outside.
__global__ void pad2d(float* __restrict__ out, const float* __restrict__ src,
                      int outR, int outC, int srcR, int srcC, int srcLd)
{
  int idx = blockIdx.x * 256 + threadIdx.x;
  if (idx >= outR * outC) return;
  int r = idx / outC, c = idx - r * outC;
  out[idx] = (r < srcR && c < srcC) ? src[(long long)r * srcLd + c] : 0.f;
}

__global__ void add_out(const float* __restrict__ a, const float* __restrict__ b,
                        float* __restrict__ c, int n)
{
  int i = blockIdx.x * 256 + threadIdx.x;
  if (i < n) c[i] = a[i] + b[i];
}

// depthwise conv (K=5, SAME) + SiLU; xz (MT,256) -> out (MT,128), channel offset
__global__ void dwconv_silu(const float* __restrict__ xz, const float* __restrict__ w,
                            float* __restrict__ out, int chanOff)
{
  int idx = blockIdx.x * 256 + threadIdx.x;     // (b,l,c) c fastest
  int c = idx & (DHALFc - 1);
  int tok = idx >> 7;
  int l = tok & (Lc - 1);
  int b = tok >> 11;
  float acc = 0.f;
#pragma unroll
  for (int t = 0; t < 5; ++t) {
    int ll = l + t - 2;
    if (ll >= 0 && ll < Lc)
      acc += w[c * 5 + t] * xz[((long long)b * Lc + ll) * 256 + chanOff + c];
  }
  out[(long long)tok * DHALFc + c] = acc / (1.f + expf(-acc));   // SiLU
}

// LayerNorm(last dim) with affine, fused exact GELU; in place, block per row.
__global__ void ln_gelu(float* __restrict__ H, const float* __restrict__ g,
                        const float* __restrict__ be, int cols)
{
  float* hp = H + (long long)blockIdx.x * cols;
  __shared__ float red[256];
  const int tid = threadIdx.x;
  float s = 0.f;
  for (int c = tid; c < cols; c += 256) s += hp[c];
  red[tid] = s; __syncthreads();
  for (int k = 128; k > 0; k >>= 1) { if (tid < k) red[tid] += red[tid+k]; __syncthreads(); }
  float mean = red[0] / cols; __syncthreads();
  float v = 0.f;
  for (int c = tid; c < cols; c += 256) { float dx = hp[c] - mean; v += dx * dx; }
  red[tid] = v; __syncthreads();
  for (int k = 128; k > 0; k >>= 1) { if (tid < k) red[tid] += red[tid+k]; __syncthreads(); }
  float rstd = rsqrtf(red[0] / cols + 1e-5f);
  for (int c = tid; c < cols; c += 256) {
    float xn = (hp[c] - mean) * rstd * g[c] + be[c];
    hp[c] = 0.5f * xn * (1.f + erff(xn * 0.70710678118f));       // exact GELU
  }
}

// Selective scan: threads = 16 d x 16 n per block; sequential over L.
// State reduction over N=16 via wave32 shfl_xor. xdbl has ld 64.
__global__ void selective_scan(const float* __restrict__ xc, const float* __restrict__ dtin,
                               const float* __restrict__ xdbl, const float* __restrict__ Alog,
                               const float* __restrict__ Dp, const float* __restrict__ dtb,
                               float* __restrict__ y)
{
  const int tid = threadIdx.x;
  const int n = tid & 15;
  const int d = blockIdx.x * 16 + (tid >> 4);
  const int b = blockIdx.y;
  const float A  = -expf(Alog[d * 16 + n]);
  const float Dv = Dp[d];
  const float bias = dtb[d];
  float h = 0.f;
  for (int l = 0; l < Lc; ++l) {
    long long row = (long long)b * Lc + l;
    float dt = dtin[row * DHALFc + d] + bias;                 // ref double-adds dt_bias
    float delta = (dt > 20.f) ? dt : log1pf(expf(dt));        // softplus
    float x  = xc[row * DHALFc + d];
    float Bv = xdbl[row * 64 + 16 + n];
    float Cv = xdbl[row * 64 + 32 + n];
    h = expf(delta * A) * h + (delta * x) * Bv;
    float p = h * Cv;
    p += __shfl_xor(p, 1, 16);
    p += __shfl_xor(p, 2, 16);
    p += __shfl_xor(p, 4, 16);
    p += __shfl_xor(p, 8, 16);
    if (n == 0) y[row * DHALFc + d] = p + Dv * x;
  }
}

// ---------------------------------------------------------------------------
// Host orchestration
// ---------------------------------------------------------------------------
static inline void gemm_nt(const float* A, const float* W, const float* bias, float* C,
                           int M, int N, int K, int lda, int ldw, int ldc,
                           long long sA, long long sW, long long sC, int batch, hipStream_t st)
{
  dim3 g(N / 64, M / 64, batch);
  wmma_gemm_nt<<<g, 128, 0, st>>>(A, W, bias, C, K, lda, ldw, ldc, sA, sW, sC);
}
static inline void gemm_nn(const float* A, const float* Bm, float* C,
                           int M, int N, int K, int lda, int ldb, int ldc,
                           long long sA, long long sB, long long sC, int batch, hipStream_t st)
{
  dim3 g(N / 64, M / 64, batch);
  wmma_gemm_nn<<<g, 128, 0, st>>>(A, Bm, C, K, lda, ldb, ldc, sA, sB, sC);
}

struct Scratch {
  float *QKV,*QH,*KH,*VH,*SC,*SC2,*OH,*OM,*OP,*XZ,*XC,*ZC,*XDBL,*XPW,*DTIN,*DTW,*DT,
        *YT,*SX,*CX,*CAT4,*H1,*FFN,*CAT2,*MH,*MM,*MP,*QK0H,*QK1H,*V0H,*V1H;
};

static void mamba_attention(const float* x, const float* pe,
                            const float* qkv_w, const float* qkv_b,
                            const float* ao_w, const float* ao_b,
                            const float* fw1, const float* fb1,
                            const float* fg, const float* fbe,
                            const float* fw2, const float* fb2,
                            const float* in_w, const float* cvx, const float* cvz,
                            const float* xp_w, const float* dt_w, const float* dt_b,
                            const float* Alog, const float* Dp, const float* out_w,
                            float* out, const Scratch& S, hipStream_t st)
{
  // self-attention
  gemm_nt(x, qkv_w, qkv_b, S.QKV, MT, 3*Ec, Ec, Ec, Ec, 3*Ec, 0, 0, 0, 1, st);
  rope_split<<<dim3(MT*Ec/256), 256, 0, st>>>(S.QKV, pe, S.QH, S.KH, S.VH);
  gemm_nt(S.QH, S.KH, nullptr, S.SC, Lc, Lc, DHc, DHc, DHc, Lc, LD, LD, LL, Bc*Hc, st);
  softmax_rows<<<dim3(Lc, Bc*Hc), 256, 0, st>>>(S.SC, Lc, 0.125f, Lc, LL);
  gemm_nn(S.SC, S.VH, S.OH, Lc, DHc, Lc, Lc, DHc, DHc, LL, LD, LD, Bc*Hc, st);
  merge_heads<<<dim3(MT*Ec/256), 256, 0, st>>>(S.OH, S.OM);
  gemm_nt(S.OM, ao_w, ao_b, S.OP, MT, Ec, Ec, Ec, Ec, Ec, 0, 0, 0, 1, st);
  // mamba two-path
  gemm_nt(x, in_w, nullptr, S.XZ, MT, 2*DHALFc, Ec, Ec, Ec, 2*DHALFc, 0, 0, 0, 1, st);
  dwconv_silu<<<dim3(MT*DHALFc/256), 256, 0, st>>>(S.XZ, cvx, S.XC, 0);
  dwconv_silu<<<dim3(MT*DHALFc/256), 256, 0, st>>>(S.XZ, cvz, S.ZC, DHALFc);
  // x_dbl: pad W rows 48->64 so N % 64 == 0; XDBL gets ld 64 (cols 48..63 junk-free zeros)
  pad2d<<<dim3(64*DHALFc/256), 256, 0, st>>>(S.XPW, xp_w, 64, DHALFc, 48, DHALFc, DHALFc);
  gemm_nt(S.XC, S.XPW, nullptr, S.XDBL, MT, 64, DHALFc, DHALFc, DHALFc, 64, 0, 0, 0, 1, st);
  // zero-pad K=16 -> 32 so the dt WMMA GEMM stays guard-free
  pad2d<<<dim3(MT*32/256), 256, 0, st>>>(S.DTIN, S.XDBL, MT, 32, MT, 16, 64);
  pad2d<<<dim3(DHALFc*32/256), 256, 0, st>>>(S.DTW, dt_w, DHALFc, 32, DHALFc, 16, 16);
  gemm_nt(S.DTIN, S.DTW, dt_b, S.DT, MT, DHALFc, 32, 32, 32, DHALFc, 0, 0, 0, 1, st);
  selective_scan<<<dim3(DHALFc/16, Bc), 256, 0, st>>>(S.XC, S.DT, S.XDBL, Alog, Dp, dt_b, S.YT);
  gemm_nt(S.YT, out_w, nullptr, S.SX, MT, Ec, DHALFc, DHALFc, DHALFc, Ec, 0, 0, 0, 1, st);
  gemm_nt(S.ZC, out_w, nullptr, S.CX, MT, Ec, DHALFc, DHALFc, DHALFc, Ec, 0, 0, 0, 1, st);
  // gated FFN on concat(x, o, s_x, c_x)
  copy_cols<<<dim3(MT*Ec/256), 256, 0, st>>>(S.CAT4, x,    MT, Ec, Ec, 4*Ec, 0);
  copy_cols<<<dim3(MT*Ec/256), 256, 0, st>>>(S.CAT4, S.OP, MT, Ec, Ec, 4*Ec, Ec);
  copy_cols<<<dim3(MT*Ec/256), 256, 0, st>>>(S.CAT4, S.SX, MT, Ec, Ec, 4*Ec, 2*Ec);
  copy_cols<<<dim3(MT*Ec/256), 256, 0, st>>>(S.CAT4, S.CX, MT, Ec, Ec, 4*Ec, 3*Ec);
  gemm_nt(S.CAT4, fw1, fb1, S.H1, MT, 2*Ec, 4*Ec, 4*Ec, 4*Ec, 2*Ec, 0, 0, 0, 1, st);
  ln_gelu<<<dim3(MT), 256, 0, st>>>(S.H1, fg, fbe, 2*Ec);
  gemm_nt(S.H1, fw2, fb2, S.FFN, MT, Ec, 2*Ec, 2*Ec, 2*Ec, Ec, 0, 0, 0, 1, st);
  add_out<<<dim3(MT*Ec/256), 256, 0, st>>>(x, S.FFN, out, MT*Ec);
}

extern "C" void kernel_launch(void* const* d_in, const int* in_sizes, int n_in,
                              void* d_out, int out_size, void* d_ws, size_t ws_size,
                              hipStream_t stream)
{
  (void)in_sizes; (void)n_in; (void)out_size; (void)ws_size;
  const float* desc0     = (const float*)d_in[0];
  const float* desc1     = (const float*)d_in[1];
  const float* pe0       = (const float*)d_in[2];
  const float* pe1       = (const float*)d_in[3];
  const float* qkv_w     = (const float*)d_in[4];
  const float* qkv_b     = (const float*)d_in[5];
  const float* ao_w      = (const float*)d_in[6];
  const float* ao_b      = (const float*)d_in[7];
  const float* ma_w1     = (const float*)d_in[8];
  const float* ma_b1     = (const float*)d_in[9];
  const float* ma_g      = (const float*)d_in[10];
  const float* ma_be     = (const float*)d_in[11];
  const float* ma_w2     = (const float*)d_in[12];
  const float* ma_b2     = (const float*)d_in[13];
  const float* m_in_w    = (const float*)d_in[14];
  const float* m_convx_w = (const float*)d_in[15];
  const float* m_convz_w = (const float*)d_in[16];
  const float* m_xproj_w = (const float*)d_in[17];
  const float* m_dt_w    = (const float*)d_in[18];
  const float* m_dt_b    = (const float*)d_in[19];
  const float* m_Alog    = (const float*)d_in[20];
  const float* m_D       = (const float*)d_in[21];
  const float* m_out_w   = (const float*)d_in[22];
  const float* ca_qk_w   = (const float*)d_in[23];
  const float* ca_qk_b   = (const float*)d_in[24];
  const float* ca_v_w    = (const float*)d_in[25];
  const float* ca_v_b    = (const float*)d_in[26];
  const float* ca_out_w  = (const float*)d_in[27];
  const float* ca_out_b  = (const float*)d_in[28];
  const float* ca_w1     = (const float*)d_in[29];
  const float* ca_b1     = (const float*)d_in[30];
  const float* ca_g      = (const float*)d_in[31];
  const float* ca_be     = (const float*)d_in[32];
  const float* ca_w2     = (const float*)d_in[33];
  const float* ca_b2     = (const float*)d_in[34];

  float* ws = (float*)d_ws;
  size_t off = 0;
  auto A = [&](size_t n) { float* p = ws + off; off += n; return p; };
  Scratch S;
  S.QKV  = A((size_t)MT * 3 * Ec);
  S.QH   = A((size_t)Bc * Hc * Lc * DHc);
  S.KH   = A((size_t)Bc * Hc * Lc * DHc);
  S.VH   = A((size_t)Bc * Hc * Lc * DHc);
  S.SC   = A((size_t)Bc * Hc * Lc * Lc);      // 134 MB (L2-resident)
  S.SC2  = A((size_t)Bc * Hc * Lc * Lc);      // transposed col-softmax
  S.OH   = A((size_t)Bc * Hc * Lc * DHc);
  S.OM   = A((size_t)MT * Ec);
  S.OP   = A((size_t)MT * Ec);
  S.XZ   = A((size_t)MT * 2 * DHALFc);
  S.XC   = A((size_t)MT * DHALFc);
  S.ZC   = A((size_t)MT * DHALFc);
  S.XDBL = A((size_t)MT * 64);
  S.XPW  = A((size_t)64 * DHALFc);
  S.DTIN = A((size_t)MT * 32);
  S.DTW  = A((size_t)DHALFc * 32);
  S.DT   = A((size_t)MT * DHALFc);
  S.YT   = A((size_t)MT * DHALFc);
  S.SX   = A((size_t)MT * Ec);
  S.CX   = A((size_t)MT * Ec);
  S.CAT4 = A((size_t)MT * 4 * Ec);
  S.H1   = A((size_t)MT * 2 * Ec);
  S.FFN  = A((size_t)MT * Ec);
  S.CAT2 = A((size_t)MT * 2 * Ec);
  S.MH   = A((size_t)Bc * Hc * Lc * DHc);
  S.MM   = A((size_t)MT * Ec);
  S.MP   = A((size_t)MT * Ec);
  S.QK0H = A((size_t)Bc * Hc * Lc * DHc);
  S.QK1H = A((size_t)Bc * Hc * Lc * DHc);
  S.V0H  = A((size_t)Bc * Hc * Lc * DHc);
  S.V1H  = A((size_t)Bc * Hc * Lc * DHc);
  float* D0 = A((size_t)MT * Ec);
  float* D1 = A((size_t)MT * Ec);
  float* outp = (float*)d_out;

  // ---- per-descriptor mamba-attention blocks ----
  mamba_attention(desc0, pe0, qkv_w, qkv_b, ao_w, ao_b, ma_w1, ma_b1, ma_g, ma_be,
                  ma_w2, ma_b2, m_in_w, m_convx_w, m_convz_w, m_xproj_w, m_dt_w,
                  m_dt_b, m_Alog, m_D, m_out_w, D0, S, stream);
  mamba_attention(desc1, pe1, qkv_w, qkv_b, ao_w, ao_b, ma_w1, ma_b1, ma_g, ma_be,
                  ma_w2, ma_b2, m_in_w, m_convx_w, m_convz_w, m_xproj_w, m_dt_w,
                  m_dt_b, m_Alog, m_D, m_out_w, D1, S, stream);

  // ---- cross attention ----
  const float sqk = 0.35355339059f;           // DH^-0.25
  gemm_nt(D0, ca_qk_w, ca_qk_b, S.QKV, MT, Ec, Ec, Ec, Ec, Ec, 0, 0, 0, 1, stream);
  split_heads_scale<<<dim3(MT*Ec/256), 256, 0, stream>>>(S.QKV, S.QK0H, sqk);
  gemm_nt(D1, ca_qk_w, ca_qk_b, S.QKV, MT, Ec, Ec, Ec, Ec, Ec, 0, 0, 0, 1, stream);
  split_heads_scale<<<dim3(MT*Ec/256), 256, 0, stream>>>(S.QKV, S.QK1H, sqk);
  gemm_nt(D0, ca_v_w, ca_v_b, S.QKV, MT, Ec, Ec, Ec, Ec, Ec, 0, 0, 0, 1, stream);
  split_heads_scale<<<dim3(MT*Ec/256), 256, 0, stream>>>(S.QKV, S.V0H, 1.f);
  gemm_nt(D1, ca_v_w, ca_v_b, S.QKV, MT, Ec, Ec, Ec, Ec, Ec, 0, 0, 0, 1, stream);
  split_heads_scale<<<dim3(MT*Ec/256), 256, 0, stream>>>(S.QKV, S.V1H, 1.f);

  // sim = qk0 @ qk1^T  per (b,h)
  gemm_nt(S.QK0H, S.QK1H, nullptr, S.SC, Lc, Lc, DHc, DHc, DHc, Lc, LD, LD, LL, Bc*Hc, stream);
  // column softmax (transposed out) BEFORE in-place row softmax
  softmax_cols_T<<<dim3(Lc, Bc*Hc), 256, 0, stream>>>(S.SC, S.SC2, Lc, Lc, LL, LL);
  softmax_rows<<<dim3(Lc, Bc*Hc), 256, 0, stream>>>(S.SC, Lc, 1.f, Lc, LL);

  // m0 path -> x0
  gemm_nn(S.SC, S.V1H, S.MH, Lc, DHc, Lc, Lc, DHc, DHc, LL, LD, LD, Bc*Hc, stream);
  merge_heads<<<dim3(MT*Ec/256), 256, 0, stream>>>(S.MH, S.MM);
  gemm_nt(S.MM, ca_out_w, ca_out_b, S.MP, MT, Ec, Ec, Ec, Ec, Ec, 0, 0, 0, 1, stream);
  copy_cols<<<dim3(MT*Ec/256), 256, 0, stream>>>(S.CAT2, D0,   MT, Ec, Ec, 2*Ec, 0);
  copy_cols<<<dim3(MT*Ec/256), 256, 0, stream>>>(S.CAT2, S.MP, MT, Ec, Ec, 2*Ec, Ec);
  gemm_nt(S.CAT2, ca_w1, ca_b1, S.H1, MT, 2*Ec, 2*Ec, 2*Ec, 2*Ec, 2*Ec, 0, 0, 0, 1, stream);
  ln_gelu<<<dim3(MT), 256, 0, stream>>>(S.H1, ca_g, ca_be, 2*Ec);
  gemm_nt(S.H1, ca_w2, ca_b2, S.FFN, MT, Ec, 2*Ec, 2*Ec, 2*Ec, Ec, 0, 0, 0, 1, stream);
  add_out<<<dim3(MT*Ec/256), 256, 0, stream>>>(D0, S.FFN, outp, MT*Ec);

  // m1 path -> x1 (uses transposed column-softmax so it is a plain NN GEMM)
  gemm_nn(S.SC2, S.V0H, S.MH, Lc, DHc, Lc, Lc, DHc, DHc, LL, LD, LD, Bc*Hc, stream);
  merge_heads<<<dim3(MT*Ec/256), 256, 0, stream>>>(S.MH, S.MM);
  gemm_nt(S.MM, ca_out_w, ca_out_b, S.MP, MT, Ec, Ec, Ec, Ec, Ec, 0, 0, 0, 1, stream);
  copy_cols<<<dim3(MT*Ec/256), 256, 0, stream>>>(S.CAT2, D1,   MT, Ec, Ec, 2*Ec, 0);
  copy_cols<<<dim3(MT*Ec/256), 256, 0, stream>>>(S.CAT2, S.MP, MT, Ec, Ec, 2*Ec, Ec);
  gemm_nt(S.CAT2, ca_w1, ca_b1, S.H1, MT, 2*Ec, 2*Ec, 2*Ec, 2*Ec, 2*Ec, 0, 0, 0, 1, stream);
  ln_gelu<<<dim3(MT), 256, 0, stream>>>(S.H1, ca_g, ca_be, 2*Ec);
  gemm_nt(S.H1, ca_w2, ca_b2, S.FFN, MT, Ec, 2*Ec, 2*Ec, 2*Ec, Ec, 0, 0, 0, 1, stream);
  add_out<<<dim3(MT*Ec/256), 256, 0, stream>>>(D1, S.FFN, outp + (size_t)MT*Ec, MT*Ec);
}
